// lpRNN_21577915695702
// MI455X (gfx1250) — matrix operand order repaired and checked by reading, not compiled, not run on hardware
//
#include <hip/hip_runtime.h>

#define T_STEPS 512
#define BATCH   32
#define HID     1024
#define INP     1024
#define NWG     64      // workgroups in the persistent recurrence kernel
#define NTILE   16      // output columns per workgroup

typedef __attribute__((ext_vector_type(16))) __bf16 v16bf;
typedef __attribute__((ext_vector_type(8)))  __bf16 bf16x8;
typedef __attribute__((ext_vector_type(8)))  float  v8f;

union AFrag { v16bf v; bf16x8 h[2]; };

// Hardware f32 -> bf16 convert (RNE on gfx1250); avoids integer bit-twiddling.
__device__ __forceinline__ __bf16 f2bf(float f) { return (__bf16)f; }

// ---------------------------------------------------------------------------
// Init: zero grid-barrier counters, convert initial hidden state to bf16.
// ---------------------------------------------------------------------------
__global__ void lp_init(const float* __restrict__ hidden,
                        __bf16* __restrict__ hbf,
                        unsigned* __restrict__ cnt) {
  int i = blockIdx.x * blockDim.x + threadIdx.x;
  if (i < BATCH * HID) hbf[i] = f2bf(hidden[i]);
  if (i < T_STEPS) cnt[i] = 0u;
}

// ---------------------------------------------------------------------------
// x_proj GEMM: out[m,n] = sum_k X[m,k]*Wih[n,k] + bias_ih[n] + bias_hh[n]
// M = T*B = 16384, N = K = 1024. 64x64 tile per workgroup, 8 waves.
// Results go straight into d_out's outputs region (read+overwritten later).
// ---------------------------------------------------------------------------
__global__ __launch_bounds__(256) void lp_xproj(const float* __restrict__ x,
                                                const float* __restrict__ wih,
                                                const float* __restrict__ bih,
                                                const float* __restrict__ bhh,
                                                float* __restrict__ out) {
  __shared__ __bf16 sA[64][32];   // [m][k]
  __shared__ __bf16 sB[64][32];   // [n][k]  (row-major weight slice)
  const int tid  = threadIdx.x;
  const int lane = tid & 31;
  const int wave = tid >> 5;                 // 0..7
  const int nblks = HID / 64;
  const int m0 = (blockIdx.x / nblks) * 64;
  const int n0 = (blockIdx.x % nblks) * 64;
  const int mt  = wave & 3;                  // M sub-tile 0..3
  const int nt0 = (wave >> 2) * 2;           // two N sub-tiles per wave
  const int col = lane & 15;
  const int hi  = lane >> 4;
  const int lr   = tid >> 2;                 // staging row 0..63
  const int loff = (tid & 3) * 8;            // staging col offset

  v8f c0 = {}; v8f c1 = {};
  for (int k0 = 0; k0 < INP; k0 += 32) {
    const float* ag = x   + (size_t)(m0 + lr) * INP + k0 + loff;
    const float* bg = wih + (size_t)(n0 + lr) * INP + k0 + loff;
#pragma unroll
    for (int j = 0; j < 8; ++j) {
      sA[lr][loff + j] = f2bf(ag[j]);
      sB[lr][loff + j] = f2bf(bg[j]);
    }
    __syncthreads();

    AFrag a, b0, b1;
    const __bf16* ar = &sA[mt * 16 + col][0];
    a.h[0] = *(const bf16x8*)(ar + hi * 16);         // K {0-7}/{16-23}
    a.h[1] = *(const bf16x8*)(ar + hi * 16 + 8);     // K {8-15}/{24-31}
    const __bf16* br0 = &sB[nt0 * 16 + col][0];
    b0.h[0] = *(const bf16x8*)(br0 + hi * 16);
    b0.h[1] = *(const bf16x8*)(br0 + hi * 16 + 8);
    const __bf16* br1 = &sB[(nt0 + 1) * 16 + col][0];
    b1.h[0] = *(const bf16x8*)(br1 + hi * 16);
    b1.h[1] = *(const bf16x8*)(br1 + hi * 16 + 8);

    c0 = __builtin_amdgcn_wmma_f32_16x16x32_bf16(false, a.v, false, b0.v,
                                                 (short)0, c0, false, false);
    c1 = __builtin_amdgcn_wmma_f32_16x16x32_bf16(false, a.v, false, b1.v,
                                                 (short)0, c1, false, false);
    __syncthreads();
  }

#pragma unroll
  for (int s = 0; s < 2; ++s) {
    int n = n0 + (nt0 + s) * 16 + col;
    float bias = bih[n] + bhh[n];
    v8f c = s ? c1 : c0;
#pragma unroll
    for (int v = 0; v < 8; ++v) {
      int m = m0 + mt * 16 + v + 8 * hi;
      out[(size_t)m * HID + n] = c[v] + bias;
    }
  }
}

// ---------------------------------------------------------------------------
// Persistent recurrence kernel. 64 workgroups x 128 threads (4 waves).
// Each workgroup owns 16 output columns; its bf16 W_hh slice lives in LDS
// for all 512 steps. Waves: (mt = M tile, kh = K half) with LDS reduction.
// Hidden state exchanged via a double-buffered bf16 global buffer + one
// grid barrier (atomic counter per step) per timestep. Two interleaved
// accumulators halve the dependent-WMMA chain on the sequential path.
// ---------------------------------------------------------------------------
__global__ __launch_bounds__(128) void lp_rnn(const float* __restrict__ hidden0,
                                              const float* __restrict__ whh,
                                              const float* __restrict__ ret,
                                              float* __restrict__ out,
                                              __bf16* __restrict__ hbf,
                                              unsigned* __restrict__ cnt) {
  __shared__ __bf16 sW[NTILE][HID];          // 32 KB: W_hh slice, bf16
  __shared__ float  sRed[2][8][32];          // partial-C reduction

  const int tid  = threadIdx.x;
  const int lane = tid & 31;
  const int wave = tid >> 5;
  const int mt = wave & 1;                   // batch-row tile (0..15 / 16..31)
  const int kh = wave >> 1;                  // K half (0..511 / 512..1023)
  const int col = lane & 15;
  const int hi  = lane >> 4;
  const int n0 = blockIdx.x * NTILE;
  const int n  = n0 + col;

  // Stage bf16 W_hh slice once (16 rows x 1024, 128 elems/thread).
  {
    const int r  = tid >> 3;
    const int cb = (tid & 7) * 128;
    const float* wrow = whh + (size_t)(n0 + r) * HID + cb;
    for (int j = 0; j < 128; ++j) sW[r][cb + j] = f2bf(wrow[j]);
  }

  const float rn = ret[n];
  float hprev[8];
#pragma unroll
  for (int v = 0; v < 8; ++v) {
    int b = mt * 16 + v + 8 * hi;
    hprev[v] = hidden0[(size_t)b * HID + n];
  }
  __syncthreads();

  const int kbeg = kh * (HID / 2);
  for (int t = 0; t < T_STEPS; ++t) {
    const __bf16* hsrc = hbf + (size_t)(t & 1) * BATCH * HID;
    __bf16*       hdst = hbf + (size_t)((t + 1) & 1) * BATCH * HID;

    v8f ca = {}; v8f cb = {};
    const __bf16* ar = hsrc + (size_t)(mt * 16 + col) * HID;
#pragma unroll 4
    for (int k0 = kbeg; k0 < kbeg + HID / 2; k0 += 64) {
      AFrag a0, b0, a1, b1;
      a0.h[0] = *(const bf16x8*)(ar + k0 + hi * 16);
      a0.h[1] = *(const bf16x8*)(ar + k0 + 8 + hi * 16);
      a1.h[0] = *(const bf16x8*)(ar + k0 + 32 + hi * 16);
      a1.h[1] = *(const bf16x8*)(ar + k0 + 40 + hi * 16);
      const __bf16* br = &sW[col][k0 + hi * 16];
      b0.h[0] = *(const bf16x8*)(br);
      b0.h[1] = *(const bf16x8*)(br + 8);
      b1.h[0] = *(const bf16x8*)(br + 32);
      b1.h[1] = *(const bf16x8*)(br + 40);
      ca = __builtin_amdgcn_wmma_f32_16x16x32_bf16(false, a0.v, false, b0.v,
                                                   (short)0, ca, false, false);
      cb = __builtin_amdgcn_wmma_f32_16x16x32_bf16(false, a1.v, false, b1.v,
                                                   (short)0, cb, false, false);
    }
    __syncthreads();
    if (kh) {
#pragma unroll
      for (int v = 0; v < 8; ++v) sRed[mt][v][lane] = ca[v] + cb[v];
    }
    __syncthreads();
    if (!kh) {
      float* xprow = out + (size_t)t * BATCH * HID;
#pragma unroll
      for (int v = 0; v < 8; ++v) {
        float acc = ca[v] + cb[v] + sRed[mt][v][lane];
        int b = mt * 16 + v + 8 * hi;
        size_t idx = (size_t)b * HID + n;
        float hy = rn * hprev[v] + (1.f - rn) * fmaxf(acc + xprow[idx], 0.f);
        hprev[v] = hy;
        xprow[idx] = hy;                     // overwrite x_proj with output
        hdst[idx] = f2bf(hy);                // publish bf16 hidden state
        if (t == T_STEPS - 1)
          out[(size_t)T_STEPS * BATCH * HID + idx] = hy;   // hidden_final
      }
      if (t + 1 < T_STEPS)
        __builtin_prefetch(xprow + BATCH * HID + (size_t)(mt * 16 + 8 * hi) * HID + n, 0, 0);
    }

    // Grid barrier: all writes visible, then arrive + spin on per-step counter.
    __threadfence();
    __syncthreads();
    if (tid == 0) {
      atomicAdd(&cnt[t], 1u);
      volatile unsigned* p = cnt + t;
      while (*p < NWG) __builtin_amdgcn_s_sleep(1);
    }
    __syncthreads();
  }
}

extern "C" void kernel_launch(void* const* d_in, const int* in_sizes, int n_in,
                              void* d_out, int out_size, void* d_ws, size_t ws_size,
                              hipStream_t stream) {
  (void)in_sizes; (void)n_in; (void)out_size; (void)ws_size;
  const float* input  = (const float*)d_in[0];   // [T,B,I]
  const float* hidden = (const float*)d_in[1];   // [B,H]
  const float* wih    = (const float*)d_in[2];   // [H,I]
  const float* whh    = (const float*)d_in[3];   // [H,H]
  const float* bih    = (const float*)d_in[4];   // [H]
  const float* bhh    = (const float*)d_in[5];   // [H]
  const float* ret    = (const float*)d_in[6];   // [H]
  float* out = (float*)d_out;                    // [T,B,H] outputs + [B,H] final

  unsigned* cnt = (unsigned*)d_ws;                       // 512 counters
  __bf16*   hbf = (__bf16*)((char*)d_ws + 4096);         // 2 x [B,H] bf16

  lp_init<<<(BATCH * HID + 255) / 256, 256, 0, stream>>>(hidden, hbf, cnt);

  const int gemm_blocks = (T_STEPS * BATCH / 64) * (HID / 64); // 4096
  lp_xproj<<<gemm_blocks, 256, 0, stream>>>(input, wih, bih, bhh, out);

  lp_rnn<<<NWG, 128, 0, stream>>>(hidden, whh, ret, out, hbf, cnt);
}
